// GeometricReasoningOriginalImpl_30442728194637
// MI455X (gfx1250) — matrix operand order repaired and checked by reading, not compiled, not run on hardware
//
#include <hip/hip_runtime.h>
#include <hip/hip_bf16.h>
#include <math.h>

typedef __attribute__((ext_vector_type(16))) __bf16 v16bf;
typedef __attribute__((ext_vector_type(8)))  __bf16 v8bf;
typedef __attribute__((ext_vector_type(8)))  float  v8f;
typedef __attribute__((ext_vector_type(2)))  float  v2f;

#define B_    2
#define S_    1024
#define CS_   1024
#define H_    32
#define DPROJ 480     // H*12 + H*3*M
#define NROT  288     // H*6 + H*3*M

// ---------------------------------------------------------------------------
// LayerNorm: per row of s (B*S rows of CS_), write bf16 normalized activations
// ---------------------------------------------------------------------------
__global__ __launch_bounds__(256) void ln_kernel(const float* __restrict__ s,
                                                 const float* __restrict__ w,
                                                 __bf16* __restrict__ nsb) {
  int row = blockIdx.x;
  const float* x = s + (size_t)row * CS_;
  __shared__ float red[256];
  float sum = 0.f, sum2 = 0.f;
  for (int c = threadIdx.x; c < CS_; c += 256) { float v = x[c]; sum += v; sum2 += v*v; }
  red[threadIdx.x] = sum; __syncthreads();
  for (int o = 128; o > 0; o >>= 1) { if (threadIdx.x < o) red[threadIdx.x] += red[threadIdx.x+o]; __syncthreads(); }
  float mu = red[0] * (1.0f / CS_); __syncthreads();
  red[threadIdx.x] = sum2; __syncthreads();
  for (int o = 128; o > 0; o >>= 1) { if (threadIdx.x < o) red[threadIdx.x] += red[threadIdx.x+o]; __syncthreads(); }
  float var = red[0] * (1.0f / CS_) - mu * mu;
  float rstd = rsqrtf(var + 1e-5f);
  for (int c = threadIdx.x; c < CS_; c += 256)
    nsb[(size_t)row * CS_ + c] = (__bf16)((x[c] - mu) * rstd * w[c]);
}

// ---------------------------------------------------------------------------
// fp32 -> bf16 conversion
// ---------------------------------------------------------------------------
__global__ void cvt_bf16(const float* __restrict__ in, __bf16* __restrict__ out, int n) {
  int i = blockIdx.x * 256 + threadIdx.x;
  if (i < n) out[i] = (__bf16)in[i];
}

// ---------------------------------------------------------------------------
// bf16 WMMA GEMM:  C[M,N] = A[M,K] * B[N,K]^T   (both row-major, K contiguous)
// One wave per 16x16 tile; A and B fragments both in contiguous K order (lane half h holds k = 16h..16h+15).
// ---------------------------------------------------------------------------
__global__ __launch_bounds__(256) void gemm_bf16(const __bf16* __restrict__ A,
                                                 const __bf16* __restrict__ Bm,
                                                 float* __restrict__ C,
                                                 int M, int N, int K, int tilesM) {
  int lane = threadIdx.x & 31;
  int wave = threadIdx.x >> 5;
  int tile = blockIdx.x * 8 + wave;
  int tm = tile % tilesM;
  int tn = tile / tilesM;
  int ln   = lane & 15;
  int half = lane >> 4;
  int mrow = tm * 16 + ln;
  int ncol = tn * 16 + ln;
  int kba = half ? 16 : 0;   // A: lanes 16-31 hold K=16..31
  int kbb = half ? 16 : 0;   // B: lanes 16-31 hold K=16..31
  const __bf16* arow = A + (size_t)mrow * K;
  const __bf16* brow = Bm + (size_t)ncol * K;
  v8f acc = {};
  for (int k0 = 0; k0 < K; k0 += 32) {
    if (k0 + 64 < K) __builtin_prefetch(arow + k0 + 64, 0, 1);
    v8bf alo = *(const v8bf*)(arow + k0 + kba);        // K = kb..kb+7
    v8bf ahi = *(const v8bf*)(arow + k0 + kba + 8);    // K = kb+8..kb+15
    v16bf a;
#pragma unroll
    for (int e = 0; e < 8; ++e) { a[e] = alo[e]; a[e + 8] = ahi[e]; }
    v16bf b = *(const v16bf*)(brow + k0 + kbb);        // K = kb..kb+15
    acc = __builtin_amdgcn_wmma_f32_16x16x32_bf16(false, a, false, b, (short)0, acc,
                                                  false, false);
  }
  int mbase = tm * 16 + (half ? 8 : 0);
#pragma unroll
  for (int r = 0; r < 8; ++r)
    C[(size_t)(mbase + r) * N + tn * 16 + ln] = acc[r];
}

// ---------------------------------------------------------------------------
// Geometry: apply rotations, build packed WMMA feature vectors
//   qdistf[b,h,s,8] = [ qx,qy,qz,|q|^2, 1,0,0,0 ]
//   kdistf[b,h,s,8] = [-2kx,-2ky,-2kz, 1, |k|^2, 0,0,0]
//   qrotf/krotf/vvec[b,h,s,4] = [x,y,z,0]
// ---------------------------------------------------------------------------
__global__ void geom_kernel(const float* __restrict__ proj, const float* __restrict__ rot,
                            const float* __restrict__ trans,
                            float* __restrict__ qdistf, float* __restrict__ kdistf,
                            float* __restrict__ qrotf,  float* __restrict__ krotf,
                            float* __restrict__ vvec) {
  int idx = blockIdx.x * blockDim.x + threadIdx.x;
  if (idx >= B_ * S_ * 160) return;
  int j  = idx % 160;
  int bs = idx / 160;
  int b = bs / S_, s = bs % S_;
  const float* R = rot + (size_t)bs * 9;
  if (j < 96) {
    const float* v = proj + (size_t)bs * DPROJ + j * 3;
    float x = v[0], y = v[1], z = v[2];
    float r0 = R[0]*x + R[1]*y + R[2]*z;
    float r1 = R[3]*x + R[4]*y + R[5]*z;
    float r2 = R[6]*x + R[7]*y + R[8]*z;
    int h = j & 31;
    float* dst = (j < 32 ? qrotf : (j < 64 ? krotf : vvec)) +
                 (((size_t)b * H_ + h) * S_ + s) * 4;
    dst[0] = r0; dst[1] = r1; dst[2] = r2; dst[3] = 0.f;
  } else {
    int jd = j - 96;
    const float* v = proj + (size_t)bs * DPROJ + NROT + jd * 3;
    const float* T = trans + (size_t)bs * 3;
    float x = v[0], y = v[1], z = v[2];
    float p0 = R[0]*x + R[1]*y + R[2]*z + T[0];
    float p1 = R[3]*x + R[4]*y + R[5]*z + T[1];
    float p2 = R[6]*x + R[7]*y + R[8]*z + T[2];
    float n2 = p0*p0 + p1*p1 + p2*p2;
    int h = jd & 31;
    if (jd < 32) {
      float* d = qdistf + (((size_t)b * H_ + h) * S_ + s) * 8;
      d[0]=p0; d[1]=p1; d[2]=p2; d[3]=n2; d[4]=1.f; d[5]=0.f; d[6]=0.f; d[7]=0.f;
    } else {
      float* d = kdistf + (((size_t)b * H_ + h) * S_ + s) * 8;
      d[0]=-2.f*p0; d[1]=-2.f*p1; d[2]=-2.f*p2; d[3]=1.f; d[4]=n2; d[5]=0.f; d[6]=0.f; d[7]=0.f;
    }
  }
}

// ---------------------------------------------------------------------------
// Flash-style attention. One wave = 16 query rows; block = 8 waves = 128 rows.
// Whole K-range for (b,h) staged in LDS via GLOBAL_LOAD_ASYNC_TO_LDS_B128
// (ASYNCcnt path: data goes straight to LDS, no VGPR round trip; INST_OFFSET
// applies to both LDS and global address, so one base covers each 32B record).
// Score tiles via 3x f32 WMMA (16x16x4). Online softmax: row-max via 16-lane
// butterfly; l and attn*V kept as per-lane partials, reduced once at the end.
// ---------------------------------------------------------------------------
__global__ __launch_bounds__(256) void attn_kernel(
    const float* __restrict__ qdistf, const float* __restrict__ kdistf,
    const float* __restrict__ qrotf,  const float* __restrict__ krotf,
    const float* __restrict__ vvec,
    const float* __restrict__ dist_scale, const float* __restrict__ rot_scale,
    const unsigned char* __restrict__ affine_mask,
    const int* __restrict__ seq_id, const int* __restrict__ chain_id,
    float* __restrict__ attn_out) {
  __shared__ float lds_kd[S_ * 8];   // 32 KB, 16B-aligned records
  __shared__ float lds_kr[S_ * 4];   // 16 KB
  __shared__ float lds_v [S_ * 4];   // 16 KB
  int bh = blockIdx.y;
  int b = bh >> 5, h = bh & 31;
  size_t base8 = (size_t)bh * S_ * 8;
  size_t base4 = (size_t)bh * S_ * 4;
  // --- async global -> LDS staging (tracked by ASYNCcnt) ---
  for (int k = threadIdx.x; k < S_; k += 256) {
    unsigned kd_l = (unsigned)(uintptr_t)&lds_kd[k * 8];
    unsigned kr_l = (unsigned)(uintptr_t)&lds_kr[k * 4];
    unsigned v_l  = (unsigned)(uintptr_t)&lds_v [k * 4];
    const float* kd_g = kdistf + base8 + (size_t)k * 8;
    const float* kr_g = krotf  + base4 + (size_t)k * 4;
    const float* v_g  = vvec   + base4 + (size_t)k * 4;
    asm volatile("global_load_async_to_lds_b128 %0, %1, off"
                 :: "v"(kd_l), "v"(kd_g) : "memory");
    asm volatile("global_load_async_to_lds_b128 %0, %1, off offset:16"
                 :: "v"(kd_l), "v"(kd_g) : "memory");
    asm volatile("global_load_async_to_lds_b128 %0, %1, off"
                 :: "v"(kr_l), "v"(kr_g) : "memory");
    asm volatile("global_load_async_to_lds_b128 %0, %1, off"
                 :: "v"(v_l),  "v"(v_g)  : "memory");
  }
  asm volatile("s_wait_asynccnt 0x0" ::: "memory");
  __syncthreads();

  const float inv_sqrt3 = 0.57735026918962576f;
  float dw = log1pf(expf(dist_scale[h]));
  float rw = log1pf(expf(rot_scale[h]));
  int lane = threadIdx.x & 31;
  int wave = threadIdx.x >> 5;
  int half = lane >> 4;
  int ln   = lane & 15;
  int kp   = half * 2;
  int q0   = (blockIdx.x * 8 + wave) * 16;
  int m    = q0 + ln;
  // A fragments (fixed per wave)
  v2f aq0, aq1, ar;
  {
    const float* qf = qdistf + base8 + (size_t)m * 8;
    aq0[0] = qf[kp]; aq0[1] = qf[kp + 1];
    aq1[0] = half ? 0.f : 1.f; aq1[1] = 0.f;         // dims [4..7] = [1,0,0,0]
    const float* qr = qrotf + base4 + (size_t)m * 4;
    ar[0] = qr[kp]; ar[1] = qr[kp + 1];
  }
  int seqq[8], chq[8];
#pragma unroll
  for (int r = 0; r < 8; ++r) {
    int qm = q0 + half * 8 + r;
    seqq[r] = seq_id[b * S_ + qm];
    chq[r]  = chain_id[b * S_ + qm];
  }
  float mrow[8], lrow[8], a0[8], a1[8], a2[8];
#pragma unroll
  for (int r = 0; r < 8; ++r) { mrow[r] = -3.0e38f; lrow[r] = 0.f; a0[r]=a1[r]=a2[r]=0.f; }
  const float NEGB = -3.0e38f;

  for (int t = 0; t < S_ / 16; ++t) {
    int k = t * 16 + ln;
    const float* kd = &lds_kd[k * 8];
    v2f b0, b1;
    if (half) { b0[0]=kd[2]; b0[1]=kd[3]; b1[0]=0.f;   b1[1]=0.f; }
    else      { b0[0]=kd[0]; b0[1]=kd[1]; b1[0]=kd[4]; b1[1]=0.f; }
    const float* kr = &lds_kr[k * 4];
    v2f br; br[0] = kr[kp]; br[1] = kr[kp + 1];
    v8f c = {};
    c = __builtin_amdgcn_wmma_f32_16x16x4_f32(false, aq0, false, b0, (short)0, c, false, false);
    c = __builtin_amdgcn_wmma_f32_16x16x4_f32(false, aq1, false, b1, (short)0, c, false, false);
    v8f cr = {};
    cr = __builtin_amdgcn_wmma_f32_16x16x4_f32(false, ar, false, br, (short)0, cr, false, false);
    int kk = b * S_ + k;
    int seqk = seq_id[kk], chk = chain_id[kk];
    unsigned char am = affine_mask[kk];
    float v0 = lds_v[k*4], v1 = lds_v[k*4+1], v2 = lds_v[k*4+2];
#pragma unroll
    for (int r = 0; r < 8; ++r) {
      float w = (cr[r] * rw - sqrtf(fmaxf(c[r], 0.f)) * dw) * inv_sqrt3;
      float bias = (seqq[r] == seqk) ? 1.f : 0.f;
      if (!am || (chq[r] != chk)) bias = NEGB;
      w += bias;
      float tmax = w;
      tmax = fmaxf(tmax, __shfl_xor(tmax, 1, 32));
      tmax = fmaxf(tmax, __shfl_xor(tmax, 2, 32));
      tmax = fmaxf(tmax, __shfl_xor(tmax, 4, 32));
      tmax = fmaxf(tmax, __shfl_xor(tmax, 8, 32));
      float mnew  = fmaxf(mrow[r], tmax);
      float scale = __expf(mrow[r] - mnew);
      float p     = __expf(w - mnew);
      lrow[r] = lrow[r] * scale + p;
      a0[r]   = a0[r]   * scale + p * v0;
      a1[r]   = a1[r]   * scale + p * v1;
      a2[r]   = a2[r]   * scale + p * v2;
      mrow[r] = mnew;
    }
  }
#pragma unroll
  for (int r = 0; r < 8; ++r) {
    for (int o = 1; o < 16; o <<= 1) {
      lrow[r] += __shfl_xor(lrow[r], o, 32);
      a0[r]   += __shfl_xor(a0[r],   o, 32);
      a1[r]   += __shfl_xor(a1[r],   o, 32);
      a2[r]   += __shfl_xor(a2[r],   o, 32);
    }
  }
  if (ln == 0) {
#pragma unroll
    for (int r = 0; r < 8; ++r) {
      int qm = q0 + half * 8 + r;
      float inv = (lrow[r] > 0.f) ? 1.f / lrow[r] : 0.f;
      float* o = attn_out + base4 + (size_t)qm * 4;
      o[0] = a0[r] * inv; o[1] = a1[r] * inv; o[2] = a2[r] * inv; o[3] = 0.f;
    }
  }
}

// ---------------------------------------------------------------------------
// Post: inverse rotation (R^T), query mask, pack to bf16 for final GEMM
// ---------------------------------------------------------------------------
__global__ void post_kernel(const float* __restrict__ attn_out, const float* __restrict__ rot,
                            const unsigned char* __restrict__ affine, __bf16* __restrict__ aob) {
  int idx = blockIdx.x * blockDim.x + threadIdx.x;
  if (idx >= B_ * S_ * H_) return;
  int h = idx & 31;
  int bs = idx >> 5;
  int b = bs / S_, s = bs % S_;
  const float* o = attn_out + (((size_t)b * H_ + h) * S_ + s) * 4;
  const float* R = rot + (size_t)bs * 9;
  float o0 = o[0], o1 = o[1], o2 = o[2];
  float mask = affine[bs] ? 1.f : 0.f;
  float r0 = (R[0]*o0 + R[3]*o1 + R[6]*o2) * mask;
  float r1 = (R[1]*o0 + R[4]*o1 + R[7]*o2) * mask;
  float r2 = (R[2]*o0 + R[5]*o1 + R[8]*o2) * mask;
  __bf16* d = aob + (size_t)bs * 96 + h * 3;
  d[0] = (__bf16)r0; d[1] = (__bf16)r1; d[2] = (__bf16)r2;
}

// ---------------------------------------------------------------------------
extern "C" void kernel_launch(void* const* d_in, const int* in_sizes, int n_in,
                              void* d_out, int out_size, void* d_ws, size_t ws_size,
                              hipStream_t stream) {
  const float* s_in   = (const float*)d_in[0];
  const float* rot    = (const float*)d_in[1];
  const float* trans  = (const float*)d_in[2];
  const float* ln_w   = (const float*)d_in[3];
  const float* proj_w = (const float*)d_in[4];
  const float* out_w  = (const float*)d_in[5];
  const float* dsc    = (const float*)d_in[6];
  const float* rsc    = (const float*)d_in[7];
  const unsigned char* amask = (const unsigned char*)d_in[8];  // jnp bool -> 1 byte
  const int* seq_id   = (const int*)d_in[9];
  const int* chain_id = (const int*)d_in[10];
  float* out = (float*)d_out;

  char* ws = (char*)d_ws;
  size_t off = 0;
  auto alloc = [&](size_t bytes) -> void* {
    void* p = ws + off;
    off += (bytes + 255) & ~(size_t)255;
    return p;
  };
  __bf16* nsb   = (__bf16*)alloc((size_t)B_*S_*CS_ * 2);
  __bf16* pwb   = (__bf16*)alloc((size_t)DPROJ*CS_ * 2);
  __bf16* owb   = (__bf16*)alloc((size_t)CS_*96 * 2);
  float*  proj  = (float*) alloc((size_t)B_*S_*DPROJ * 4);
  float*  qdf   = (float*) alloc((size_t)B_*H_*S_*8 * 4);
  float*  kdf   = (float*) alloc((size_t)B_*H_*S_*8 * 4);
  float*  qrf   = (float*) alloc((size_t)B_*H_*S_*4 * 4);
  float*  krf   = (float*) alloc((size_t)B_*H_*S_*4 * 4);
  float*  vv    = (float*) alloc((size_t)B_*H_*S_*4 * 4);
  float*  aout  = (float*) alloc((size_t)B_*H_*S_*4 * 4);
  __bf16* aob   = (__bf16*)alloc((size_t)B_*S_*96 * 2);
  (void)ws_size; (void)n_in; (void)in_sizes; (void)out_size;

  // 1. LayerNorm -> bf16
  ln_kernel<<<B_*S_, 256, 0, stream>>>(s_in, ln_w, nsb);
  // 2. weight conversions
  cvt_bf16<<<(DPROJ*CS_ + 255)/256, 256, 0, stream>>>(proj_w, pwb, DPROJ*CS_);
  cvt_bf16<<<(CS_*96 + 255)/256, 256, 0, stream>>>(out_w, owb, CS_*96);
  // 3. proj = ns @ proj_w.T   (M=2048, N=480, K=1024) -> 128x30 tiles
  gemm_bf16<<<(128*30)/8, 256, 0, stream>>>(nsb, pwb, proj, B_*S_, DPROJ, CS_, 128);
  // 4. geometry / feature packing
  geom_kernel<<<(B_*S_*160 + 255)/256, 256, 0, stream>>>(proj, rot, trans, qdf, kdf, qrf, krf, vv);
  // 5. attention (flash-style, f32 WMMA scores, async LDS staging)
  attn_kernel<<<dim3(S_/128, B_*H_), 256, 0, stream>>>(qdf, kdf, qrf, krf, vv, dsc, rsc,
                                                       amask, seq_id, chain_id, aout);
  // 6. inverse rotation + mask -> bf16
  post_kernel<<<(B_*S_*H_ + 255)/256, 256, 0, stream>>>(aout, rot, amask, aob);
  // 7. out = ao @ out_w.T   (M=2048, N=1024, K=96) -> 128x64 tiles
  gemm_bf16<<<(128*64)/8, 256, 0, stream>>>(aob, owb, out, B_*S_, CS_, 96, 128);
}
